// superatomAttention_38534446579799
// MI455X (gfx1250) — matrix operand order repaired and checked by reading, not compiled, not run on hardware
//
#include <hip/hip_runtime.h>
#include <hip/hip_bf16.h>
#include <math.h>

#define D_DIM 256

typedef __bf16 bf16_t;
typedef __attribute__((ext_vector_type(16))) bf16_t v16bf;
typedef __attribute__((ext_vector_type(8)))  float  v8f;

#define WMMA_BF16(a, b, c) \
  __builtin_amdgcn_wmma_f32_16x16x32_bf16(false, (a), false, (b), (short)0, (c), false, false)

// ---------------- helpers ----------------

static __device__ __forceinline__ unsigned short f2bf(float f) {
  unsigned int u = __float_as_uint(f);
  if ((u & 0x7F800000u) == 0x7F800000u) return (unsigned short)(u >> 16); // inf/nan
  unsigned int r = u + 0x7FFFu + ((u >> 16) & 1u);                        // RNE
  return (unsigned short)(r >> 16);
}

static __device__ __forceinline__ float wave_sum(float x) {
#pragma unroll
  for (int off = 16; off > 0; off >>= 1) x += __shfl_xor(x, off, 32);
  return x;
}

static __device__ __forceinline__ float wave_max(float x) {
#pragma unroll
  for (int off = 16; off > 0; off >>= 1) x = fmaxf(x, __shfl_xor(x, off, 32));
  return x;
}

union ABits { uint4 u; bf16_t h[8]; };

// A-matrix fragment (16x32 bf16, row-major source [rows, 256]).
// layout: lane L<16 -> row m0+L, elems 0..15 = K k0..k0+15
//         lane L>=16 -> row m0+(L-16), elems 0..15 = K k0+16..k0+31
static __device__ __forceinline__ v16bf load_a_frag(const unsigned short* A, int row_base,
                                                    int k0, int lane) {
  int row = row_base + (lane & 15);
  int kb  = k0 + ((lane & 16) ? 16 : 0);
  const unsigned short* p = A + (size_t)row * D_DIM + kb;
  ABits lo, hi;
  lo.u = *(const uint4*)(p);
  hi.u = *(const uint4*)(p + 8);
  v16bf v;
#pragma unroll
  for (int i = 0; i < 8; ++i) { v[i] = lo.h[i]; v[i + 8] = hi.h[i]; }
  return v;
}

// B-matrix fragment (32x16 bf16). B[k][c] = W[d0+c][k], W row-major [Dout, 256].
// lane c<16 -> column c, K k0..k0+15 ; lane c+16 -> column c, K k0+16..k0+31
static __device__ __forceinline__ v16bf load_b_frag(const unsigned short* W, int d_base,
                                                    int k0, int lane) {
  int row = d_base + (lane & 15);
  int kb  = k0 + ((lane & 16) ? 16 : 0);
  const unsigned short* p = W + (size_t)row * D_DIM + kb;
  ABits lo, hi;
  lo.u = *(const uint4*)(p);
  hi.u = *(const uint4*)(p + 8);
  v16bf v;
#pragma unroll
  for (int i = 0; i < 8; ++i) { v[i] = lo.h[i]; v[i + 8] = hi.h[i]; }
  return v;
}

// ---------------- kernels ----------------

// generic f32 -> bf16 conversion
__global__ void k_f32_to_bf16(const float* __restrict__ src, unsigned short* __restrict__ dst,
                              int n) {
  int i = blockIdx.x * blockDim.x + threadIdx.x;
  if (i < n) dst[i] = f2bf(src[i]);
}

// fold BatchNorm into attend weights: Wfold = W * g*rsqrt(var+eps), biasfold per row
__global__ void k_fold_attend(const float* __restrict__ attend_w, const float* __restrict__ attend_b,
                              const float* __restrict__ g, const float* __restrict__ beta,
                              const float* __restrict__ mean, const float* __restrict__ var,
                              unsigned short* __restrict__ wfold, float* __restrict__ biasfold) {
  int i = blockIdx.x * blockDim.x + threadIdx.x;   // 256*256 threads
  int d = i >> 8, k = i & 255;
  float s = g[d] * rsqrtf(var[d] + 1e-6f);
  wfold[i] = f2bf(attend_w[i] * s);
  if (k == 0) biasfold[d] = beta[d] + (attend_b[d] - mean[d]) * s;
}

// one wave per molecule: s_mol[m] = align_w[0:256] . superatom[m], segment start via binary search
__global__ void k_mol_prep(const float* __restrict__ superatom, const float* __restrict__ align_w,
                           const int* __restrict__ mol_index, float* __restrict__ s_mol,
                           int* __restrict__ seg, int Mn, int Nn) {
  int wave = (blockIdx.x * blockDim.x + threadIdx.x) >> 5;
  int lane = threadIdx.x & 31;
  if (wave >= Mn) return;
  const float* row = superatom + (size_t)wave * D_DIM;
  float p = 0.f;
#pragma unroll
  for (int h = 0; h < 2; ++h) {
    int base = h * 128 + 4 * lane;
    float4 a = *(const float4*)(row + base);
    float4 w = *(const float4*)(align_w + base);
    p += a.x * w.x + a.y * w.y + a.z * w.z + a.w * w.w;
  }
  p = wave_sum(p);
  if (lane == 0) {
    s_mol[wave] = p;
    int lo = 0, hi = Nn;
    while (lo < hi) { int mid = (lo + hi) >> 1; if (mol_index[mid] < wave) lo = mid + 1; else hi = mid; }
    seg[wave] = lo;
    if (wave == 0) seg[Mn] = Nn;
  }
}

// one wave per atom: score = leaky_relu(s_mol[mol] + align_w[256:512].atom + b)
__global__ void k_align(const float* __restrict__ atom, const float* __restrict__ align_w,
                        const float* __restrict__ align_b, const float* __restrict__ s_mol,
                        const int* __restrict__ mol_index, float* __restrict__ score, int Nn) {
  int wave = (blockIdx.x * blockDim.x + threadIdx.x) >> 5;
  int lane = threadIdx.x & 31;
  if (wave >= Nn) return;
  const float* row = atom + (size_t)wave * D_DIM;
  const float* aw  = align_w + D_DIM;
  float p = 0.f;
#pragma unroll
  for (int h = 0; h < 2; ++h) {
    int base = h * 128 + 4 * lane;
    float4 a = *(const float4*)(row + base);
    float4 w = *(const float4*)(aw + base);
    p += a.x * w.x + a.y * w.y + a.z * w.z + a.w * w.w;
  }
  p = wave_sum(p);
  if (lane == 0) {
    float s = p + s_mol[mol_index[wave]] + align_b[0];
    score[wave] = (s > 0.f) ? s : 0.01f * s;
  }
}

// one wave per molecule: segment softmax -> attention weights, then weighted sum of atom rows
__global__ void k_seg_ctx(const float* __restrict__ score, const int* __restrict__ seg,
                          const float* __restrict__ atom, float* __restrict__ attw,
                          float* __restrict__ wsum, unsigned short* __restrict__ ctxraw, int Mn) {
  int wave = (blockIdx.x * blockDim.x + threadIdx.x) >> 5;
  int lane = threadIdx.x & 31;
  if (wave >= Mn) return;
  int s = seg[wave], e = seg[wave + 1];

  float mx = -3.402823e38f;
  for (int i = s + lane; i < e; i += 32) mx = fmaxf(mx, score[i]);
  mx = wave_max(mx);

  float sum = 0.f;
  for (int i = s + lane; i < e; i += 32) sum += expf(score[i] - mx);
  sum = wave_sum(sum);
  float inv = 1.f / (sum + 1e-8f);

  for (int i = s + lane; i < e; i += 32) attw[i] = expf(score[i] - mx) * inv;

  float4 acc0 = make_float4(0.f, 0.f, 0.f, 0.f);
  float4 acc1 = make_float4(0.f, 0.f, 0.f, 0.f);
  for (int i = s; i < e; ++i) {
    float w = expf(score[i] - mx) * inv;   // broadcast scalar per atom
    const float* row = atom + (size_t)i * D_DIM;
    float4 a0 = *(const float4*)(row + 4 * lane);
    float4 a1 = *(const float4*)(row + 128 + 4 * lane);
    acc0.x += w * a0.x; acc0.y += w * a0.y; acc0.z += w * a0.z; acc0.w += w * a0.w;
    acc1.x += w * a1.x; acc1.y += w * a1.y; acc1.z += w * a1.z; acc1.w += w * a1.w;
  }
  if (lane == 0) wsum[wave] = sum * inv;

  unsigned short* dst = ctxraw + (size_t)wave * D_DIM;
  uint2 p0, p1;
  p0.x = (unsigned)f2bf(acc0.x) | ((unsigned)f2bf(acc0.y) << 16);
  p0.y = (unsigned)f2bf(acc0.z) | ((unsigned)f2bf(acc0.w) << 16);
  p1.x = (unsigned)f2bf(acc1.x) | ((unsigned)f2bf(acc1.y) << 16);
  p1.y = (unsigned)f2bf(acc1.z) | ((unsigned)f2bf(acc1.w) << 16);
  *(uint2*)(dst + 4 * lane) = p0;
  *(uint2*)(dst + 128 + 4 * lane) = p1;
}

// context = elu(ctxraw @ Wfold^T + wsum*biasfold), one wave per 16x16 tile, WMMA bf16
__global__ void k_ctx_gemm(const unsigned short* __restrict__ Abf, const unsigned short* __restrict__ Wbf,
                           const float* __restrict__ biasfold, const float* __restrict__ wsum,
                           unsigned short* __restrict__ ctx_bf, int Mn) {
  int wave = (blockIdx.x * blockDim.x + threadIdx.x) >> 5;
  int lane = threadIdx.x & 31;
  int ntiles = (Mn >> 4) * 16;
  if (wave >= ntiles) return;
  int m0 = (wave >> 4) << 4;
  int d0 = (wave & 15) << 4;

  v8f acc = {0.f, 0.f, 0.f, 0.f, 0.f, 0.f, 0.f, 0.f};
#pragma unroll
  for (int k0 = 0; k0 < D_DIM; k0 += 32) {
    v16bf a = load_a_frag(Abf, m0, k0, lane);
    v16bf b = load_b_frag(Wbf, d0, k0, lane);
    acc = WMMA_BF16(a, b, acc);
  }

  int col = d0 + (lane & 15);
  int rb  = m0 + ((lane & 16) ? 8 : 0);
  float bf = biasfold[col];
#pragma unroll
  for (int j = 0; j < 8; ++j) {
    int m = rb + j;
    float v = acc[j] + wsum[m] * bf;
    v = (v > 0.f) ? v : (expf(v) - 1.f);          // ELU
    ctx_bf[(size_t)m * D_DIM + col] = f2bf(v);
  }
}

// fused GRU: six gate GEMMs (x/h paths, r/z/n) + elementwise update, one wave per 16x16 tile
__global__ void k_gru_gemm(const unsigned short* __restrict__ ctx_bf, const unsigned short* __restrict__ sup_bf,
                           const unsigned short* __restrict__ wih_bf, const unsigned short* __restrict__ whh_bf,
                           const float* __restrict__ bih, const float* __restrict__ bhh,
                           const float* __restrict__ superatom, float* __restrict__ out, int Mn) {
  int wave = (blockIdx.x * blockDim.x + threadIdx.x) >> 5;
  int lane = threadIdx.x & 31;
  int ntiles = (Mn >> 4) * 16;
  if (wave >= ntiles) return;
  int m0 = (wave >> 4) << 4;
  int d0 = (wave & 15) << 4;

  v8f axr = {0.f,0.f,0.f,0.f,0.f,0.f,0.f,0.f};
  v8f axz = axr, axn = axr, ahr = axr, ahz = axr, ahn = axr;

#pragma unroll
  for (int k0 = 0; k0 < D_DIM; k0 += 32) {
    v16bf ax = load_a_frag(ctx_bf, m0, k0, lane);
    v16bf ah = load_a_frag(sup_bf, m0, k0, lane);
    v16bf b;
    b = load_b_frag(wih_bf, d0,       k0, lane); axr = WMMA_BF16(ax, b, axr);
    b = load_b_frag(wih_bf, d0 + 256, k0, lane); axz = WMMA_BF16(ax, b, axz);
    b = load_b_frag(wih_bf, d0 + 512, k0, lane); axn = WMMA_BF16(ax, b, axn);
    b = load_b_frag(whh_bf, d0,       k0, lane); ahr = WMMA_BF16(ah, b, ahr);
    b = load_b_frag(whh_bf, d0 + 256, k0, lane); ahz = WMMA_BF16(ah, b, ahz);
    b = load_b_frag(whh_bf, d0 + 512, k0, lane); ahn = WMMA_BF16(ah, b, ahn);
  }

  int col = d0 + (lane & 15);
  int rb  = m0 + ((lane & 16) ? 8 : 0);
  float bxr = bih[col], bxz = bih[col + 256], bxn = bih[col + 512];
  float bhr = bhh[col], bhz = bhh[col + 256], bhn = bhh[col + 512];
#pragma unroll
  for (int j = 0; j < 8; ++j) {
    int m = rb + j;
    float r = 1.f / (1.f + expf(-((axr[j] + bxr) + (ahr[j] + bhr))));
    float z = 1.f / (1.f + expf(-((axz[j] + bxz) + (ahz[j] + bhz))));
    float n = tanhf((axn[j] + bxn) + r * (ahn[j] + bhn));
    float h = superatom[(size_t)m * D_DIM + col];
    out[(size_t)m * D_DIM + col] = (1.f - z) * n + z * h;
  }
}

// ---------------- host ----------------

extern "C" void kernel_launch(void* const* d_in, const int* in_sizes, int n_in,
                              void* d_out, int out_size, void* d_ws, size_t ws_size,
                              hipStream_t stream) {
  const float* superatom = (const float*)d_in[0];
  const float* atom      = (const float*)d_in[1];
  const int*   mol_index = (const int*)d_in[2];
  const float* align_w   = (const float*)d_in[3];
  const float* align_b   = (const float*)d_in[4];
  const float* attend_w  = (const float*)d_in[5];
  const float* attend_b  = (const float*)d_in[6];
  const float* bn_gamma  = (const float*)d_in[7];
  const float* bn_beta   = (const float*)d_in[8];
  const float* bn_mean   = (const float*)d_in[9];
  const float* bn_var    = (const float*)d_in[10];
  const float* gru_wih   = (const float*)d_in[11];
  const float* gru_whh   = (const float*)d_in[12];
  const float* gru_bih   = (const float*)d_in[13];
  const float* gru_bhh   = (const float*)d_in[14];

  const int Mn = in_sizes[0] / D_DIM;   // 16384
  const int Nn = in_sizes[2];           // 500000
  const int GW = 3 * D_DIM * D_DIM;     // GRU weight elements

  float* out_update = (float*)d_out;
  float* out_attw   = out_update + (size_t)Mn * D_DIM;

  // carve workspace (256B-aligned slabs)
  char* w = (char*)d_ws;
  auto carve = [&](size_t bytes) -> char* {
    char* p = w;
    w += (bytes + 255) & ~(size_t)255;
    return p;
  };
  float*          s_mol    = (float*)carve((size_t)Mn * 4);
  int*            seg      = (int*)carve((size_t)(Mn + 1) * 4);
  float*          score    = (float*)carve((size_t)Nn * 4);
  float*          wsum     = (float*)carve((size_t)Mn * 4);
  float*          biasfold = (float*)carve(D_DIM * 4);
  unsigned short* ctxraw   = (unsigned short*)carve((size_t)Mn * D_DIM * 2);
  unsigned short* ctx_bf   = (unsigned short*)carve((size_t)Mn * D_DIM * 2);
  unsigned short* sup_bf   = (unsigned short*)carve((size_t)Mn * D_DIM * 2);
  unsigned short* wfold    = (unsigned short*)carve((size_t)D_DIM * D_DIM * 2);
  unsigned short* wih_bf   = (unsigned short*)carve((size_t)GW * 2);
  unsigned short* whh_bf   = (unsigned short*)carve((size_t)GW * 2);

  const int BLK = 256;

  // bf16 conversions of GEMM operands
  k_f32_to_bf16<<<(Mn * D_DIM + BLK - 1) / BLK, BLK, 0, stream>>>(superatom, sup_bf, Mn * D_DIM);
  k_f32_to_bf16<<<(GW + BLK - 1) / BLK, BLK, 0, stream>>>(gru_wih, wih_bf, GW);
  k_f32_to_bf16<<<(GW + BLK - 1) / BLK, BLK, 0, stream>>>(gru_whh, whh_bf, GW);

  // fold BN into attend weights
  k_fold_attend<<<(D_DIM * D_DIM) / BLK, BLK, 0, stream>>>(
      attend_w, attend_b, bn_gamma, bn_beta, bn_mean, bn_var, wfold, biasfold);

  // per-molecule precompute: s_mol + segment starts
  k_mol_prep<<<(Mn + 7) / 8, BLK, 0, stream>>>(superatom, align_w, mol_index, s_mol, seg, Mn, Nn);

  // per-atom align scores
  k_align<<<(Nn + 7) / 8, BLK, 0, stream>>>(atom, align_w, align_b, s_mol, mol_index, score, Nn);

  // segment softmax + weighted sum of atom rows (-> ctxraw bf16, attention weights out)
  k_seg_ctx<<<(Mn + 7) / 8, BLK, 0, stream>>>(score, seg, atom, out_attw, wsum, ctxraw, Mn);

  // context GEMM (WMMA) + ELU
  int gemm_waves  = (Mn / 16) * 16;
  int gemm_blocks = (gemm_waves * 32 + BLK - 1) / BLK;
  k_ctx_gemm<<<gemm_blocks, BLK, 0, stream>>>(ctxraw, wfold, biasfold, wsum, ctx_bf, Mn);

  // fused GRU GEMMs (WMMA) + elementwise update
  k_gru_gemm<<<gemm_blocks, BLK, 0, stream>>>(ctx_bf, sup_bf, wih_bf, whh_bf,
                                              gru_bih, gru_bhh, superatom, out_update, Mn);
}
